// BasicBlock_47382079209891
// MI455X (gfx1250) — matrix operand order repaired and checked by reading, not compiled, not run on hardware
//
#include <hip/hip_runtime.h>
#include <hip/hip_bf16.h>
#include <cstdint>
#include <cstddef>

typedef __attribute__((ext_vector_type(16))) _Float16 v16h;
typedef __attribute__((ext_vector_type(8)))  _Float16 v8h;
typedef __attribute__((ext_vector_type(8)))  float    v8f;

union V16H { v16h v; v8h q[2]; _Float16 h[16]; };

#define B_SZ  256
#define T_LEN 336
#define H_DIM 512
#define F_LEN 96

// ---------------- generic f32 -> f16 convert ----------------
__global__ void cvt_f32_to_f16_kernel(const float* __restrict__ src,
                                      _Float16* __restrict__ dst, int n) {
  int i = blockIdx.x * blockDim.x + threadIdx.x;
  if (i < n) dst[i] = (_Float16)src[i];
}

// ---------------- conv weight repack: f32 [co][ci][tap] -> f16 [co][tap][ci] ----------------
__global__ void repack_w_kernel(const float* __restrict__ src, _Float16* __restrict__ dst,
                                int Cout, int Cin, int K) {
  int i = blockIdx.x * blockDim.x + threadIdx.x;
  int tot = Cout * Cin * K;
  if (i < tot) {
    int co  = i / (K * Cin);
    int rem = i - co * K * Cin;
    int tap = rem / Cin;
    int ci  = rem - tap * Cin;
    dst[i] = (_Float16)src[((size_t)co * Cin + ci) * K + tap];
  }
}

// ---------------- TCN dilated causal conv via WMMA ----------------
// out[b,co,t] = f( sum_{tap,ci} w[co,tap,ci] * x[b,ci,t-(K-1-tap)*d] + bias[co] )
// mode 0: linear (downsample path); 1: relu; 2: relu, then + res, then relu
// Wave computes 32(co) x 16(t); A fed by 16B vector loads, B by ds b128 loads.
template <int KTAPS, int DIL>
__global__ void __launch_bounds__(128) conv_wmma_kernel(
    const _Float16* __restrict__ x, const _Float16* __restrict__ wH,
    const float* __restrict__ bias, const _Float16* __restrict__ res,
    _Float16* __restrict__ out, int Cin, int Cout, int mode)
{
  constexpr int W = 16 + (KTAPS - 1) * DIL;   // compile-time time window
  constexpr int CSTRIDE = 520;                // 512 + 8 pad: 16B-aligned, bank-skewed
  __shared__ _Float16 sx[W * CSTRIDE];        // LDS tile, layout [t][ci]
  const int t0     = blockIdx.x * 16;
  const int b      = blockIdx.y;
  const int tstart = t0 - (KTAPS - 1) * DIL;

  // Stage input window into LDS (global reads coalesced in t), causal zero-pad.
  const int tot = Cin * W;
  for (int idx = threadIdx.x; idx < tot; idx += blockDim.x) {
    int ci = idx / W;                         // constant divisor -> mul/shift
    int tt = idx - ci * W;
    int tg = tstart + tt;
    _Float16 v = (_Float16)0.0f;
    if (tg >= 0) v = x[((size_t)b * Cin + ci) * T_LEN + tg];
    sx[tt * CSTRIDE + ci] = v;
  }
  // Hint the next time-tile into cache (global_prefetch_b8).
  if (t0 + 16 < T_LEN) {
    __builtin_prefetch(x + ((size_t)b * Cin + (threadIdx.x & 31)) * T_LEN + t0 + 16, 0, 0);
  }
  __syncthreads();

  const int lane  = threadIdx.x & 31;
  const int wave  = threadIdx.x >> 5;
  const int n     = lane & 15;      // B-matrix column -> output time
  const int khalf = lane >> 4;      // which K-half this lane holds
  const int koff  = khalf ? 8 : 0;  // A-matrix K offset per ISA layout
  const int m     = lane & 15;      // A-matrix row -> output channel

  for (int coT = wave; coT < (Cout >> 5); coT += 4) {
    const int co0 = coT * 32;
    v8f acc0 = {}, acc1 = {};
#pragma unroll
    for (int tap = 0; tap < KTAPS; ++tap) {
      const int tb = tap * DIL + n;           // position inside staged window
      const _Float16* wrow0 = wH + ((size_t)(co0 + m) * KTAPS + tap) * Cin;
      const _Float16* wrow1 = wrow0 + (size_t)16 * KTAPS * Cin;
      for (int kc = 0; kc < (Cin >> 5); ++kc) {
        V16H A0, A1, Bm;
        A0.q[0] = *(const v8h*)(wrow0 + kc * 32 + koff);
        A0.q[1] = *(const v8h*)(wrow0 + kc * 32 + 16 + koff);
        A1.q[0] = *(const v8h*)(wrow1 + kc * 32 + koff);
        A1.q[1] = *(const v8h*)(wrow1 + kc * 32 + 16 + koff);
        const _Float16* brow = &sx[tb * CSTRIDE + kc * 32 + koff];
        Bm.q[0] = *(const v8h*)(brow);
        Bm.q[1] = *(const v8h*)(brow + 16);
        acc0 = __builtin_amdgcn_wmma_f32_16x16x32_f16(false, A0.v, false, Bm.v,
                                                      (short)0, acc0, false, false);
        acc1 = __builtin_amdgcn_wmma_f32_16x16x32_f16(false, A1.v, false, Bm.v,
                                                      (short)0, acc1, false, false);
      }
    }
    const int t = t0 + n;
#pragma unroll
    for (int r = 0; r < 8; ++r) {
      int co = co0 + r + (khalf ? 8 : 0);
      float v = acc0[r] + bias[co];
      if (mode >= 1) v = fmaxf(v, 0.0f);
      if (mode == 2) {
        v += (float)res[((size_t)b * Cout + co) * T_LEN + t];
        v = fmaxf(v, 0.0f);
      }
      out[((size_t)b * Cout + co) * T_LEN + t] = (_Float16)v;
    }
#pragma unroll
    for (int r = 0; r < 8; ++r) {
      int co = co0 + 16 + r + (khalf ? 8 : 0);
      float v = acc1[r] + bias[co];
      if (mode >= 1) v = fmaxf(v, 0.0f);
      if (mode == 2) {
        v += (float)res[((size_t)b * Cout + co) * T_LEN + t];
        v = fmaxf(v, 0.0f);
      }
      out[((size_t)b * Cout + co) * T_LEN + t] = (_Float16)v;
    }
  }
}

// ---------------- generic GEMM: out[m,n] = act(sum_k A[m,k]*Bw[n,k] + bias[n]) ----------------
// KDIM compile-time: K loop fully unrolls; 16B vector operand loads on full chunks.
template <int KDIM>
__global__ void __launch_bounds__(128) gemm_wmma_kernel(
    const _Float16* __restrict__ A, const _Float16* __restrict__ Bw,
    const float* __restrict__ bias, float* __restrict__ outF,
    _Float16* __restrict__ outH, int Mdim, int Ndim, int act)
{
  constexpr int KCH = (KDIM + 31) >> 5;
  const int lane  = threadIdx.x & 31;
  const int wave  = threadIdx.x >> 5;
  const int pTile = blockIdx.x * 4 + wave;
  const int n0    = pTile * 32;
  if (n0 >= Ndim) return;
  const bool hasN1 = (n0 + 16) < Ndim;
  const int m0    = blockIdx.y * 16;
  const int n     = lane & 15;
  const int khalf = lane >> 4;
  const int koff  = khalf ? 8 : 0;
  const int m     = lane & 15;

  v8f acc0 = {}, acc1 = {};
  const _Float16* arow  = A  + (size_t)(m0 + m) * KDIM;
  const _Float16* b0row = Bw + (size_t)(n0 + n) * KDIM;
  const _Float16* b1row = hasN1 ? (b0row + (size_t)16 * KDIM) : b0row;
#pragma unroll
  for (int kc = 0; kc < KCH; ++kc) {
    V16H Av, B0, B1;
    if (kc * 32 + 31 < KDIM) {               // full chunk: vector loads
      Av.q[0] = *(const v8h*)(arow + kc * 32 + khalf * 16);
      Av.q[1] = *(const v8h*)(arow + kc * 32 + khalf * 16 + 8);
      B0.q[0] = *(const v8h*)(b0row + kc * 32 + khalf * 16);
      B0.q[1] = *(const v8h*)(b0row + kc * 32 + khalf * 16 + 8);
      B1.q[0] = *(const v8h*)(b1row + kc * 32 + khalf * 16);
      B1.q[1] = *(const v8h*)(b1row + kc * 32 + khalf * 16 + 8);
    } else {                                  // tail chunk with bounds guards
#pragma unroll
      for (int e = 0; e < 16; ++e) {
        int kk = kc * 32 + khalf * 16 + e;
        Av.h[e] = (kk < KDIM) ? arow[kk] : (_Float16)0.0f;
      }
#pragma unroll
      for (int e = 0; e < 16; ++e) {
        int kk = kc * 32 + khalf * 16 + e;
        B0.h[e] = (kk < KDIM) ? b0row[kk] : (_Float16)0.0f;
        B1.h[e] = (kk < KDIM) ? b1row[kk] : (_Float16)0.0f;
      }
    }
    acc0 = __builtin_amdgcn_wmma_f32_16x16x32_f16(false, Av.v, false, B0.v,
                                                  (short)0, acc0, false, false);
    acc1 = __builtin_amdgcn_wmma_f32_16x16x32_f16(false, Av.v, false, B1.v,
                                                  (short)0, acc1, false, false);
  }
#pragma unroll
  for (int r = 0; r < 8; ++r) {
    int row = m0 + r + (khalf ? 8 : 0);
    int col = n0 + n;
    float v = acc0[r] + (bias ? bias[col] : 0.0f);
    if (act == 1) v = fmaxf(v, 0.0f);
    outF[(size_t)row * Ndim + col] = v;
    if (outH) outH[(size_t)row * Ndim + col] = (_Float16)v;
  }
  if (hasN1) {
#pragma unroll
    for (int r = 0; r < 8; ++r) {
      int row = m0 + r + (khalf ? 8 : 0);
      int col = n0 + 16 + n;
      float v = acc1[r] + (bias ? bias[col] : 0.0f);
      if (act == 1) v = fmaxf(v, 0.0f);
      outF[(size_t)row * Ndim + col] = v;
      if (outH) outH[(size_t)row * Ndim + col] = (_Float16)v;
    }
  }
}

// ---------------- GRU gate / update / output projection ----------------
__global__ void __launch_bounds__(512) gru_gate_kernel(
    const float* __restrict__ gh, const float* __restrict__ wih,
    const float* __restrict__ bih, const float* __restrict__ fcw,
    const float* __restrict__ fcb, float* __restrict__ h,
    _Float16* __restrict__ hH, float* __restrict__ xcur,
    float* __restrict__ outSeq, int step, int outStride)
{
  __shared__ float red[512];
  const int b = blockIdx.x, j = threadIdx.x;
  const float x = xcur[b];
  const float* ghb = gh + (size_t)b * 3 * H_DIM;
  float r  = 1.0f / (1.0f + __expf(-(x * wih[j]            + bih[j]            + ghb[j])));
  float z  = 1.0f / (1.0f + __expf(-(x * wih[H_DIM + j]    + bih[H_DIM + j]    + ghb[H_DIM + j])));
  float nn = tanhf(x * wih[2 * H_DIM + j] + bih[2 * H_DIM + j] + r * ghb[2 * H_DIM + j]);
  float hp = h[(size_t)b * H_DIM + j];
  float hn = (1.0f - z) * nn + z * hp;
  h[(size_t)b * H_DIM + j]  = hn;
  hH[(size_t)b * H_DIM + j] = (_Float16)hn;
  red[j] = hn * fcw[j];
  __syncthreads();
  for (int s = 256; s > 0; s >>= 1) {
    if (j < s) red[j] += red[j + s];
    __syncthreads();
  }
  if (j == 0) {
    float o = red[0] + fcb[0];
    xcur[b] = o;
    outSeq[(size_t)b * outStride + step] = o;
  }
}

// ---------------- small helpers ----------------
__global__ void extract_hidden_kernel(const _Float16* __restrict__ act,
                                      float* __restrict__ hid, _Float16* __restrict__ hidH) {
  int i = blockIdx.x * blockDim.x + threadIdx.x;
  if (i < B_SZ * H_DIM) {
    float v = (float)act[(size_t)i * T_LEN + (T_LEN - 1)];
    hid[i] = v; hidH[i] = (_Float16)v;
  }
}

__global__ void init_h_kernel(const float* __restrict__ hid, float* __restrict__ h,
                              _Float16* __restrict__ hH) {
  int i = blockIdx.x * blockDim.x + threadIdx.x;
  if (i < B_SZ * H_DIM) { float v = hid[i]; h[i] = v; hH[i] = (_Float16)v; }
}

__global__ void f0_init_kernel(const float* __restrict__ arOut, float* __restrict__ xcur) {
  int b = blockIdx.x * blockDim.x + threadIdx.x;
  if (b < B_SZ) xcur[b] = arOut[(size_t)b * F_LEN];
}

__global__ void __launch_bounds__(512) e0_kernel(const float* __restrict__ hid,
                                                 const float* __restrict__ h2iw,
                                                 const float* __restrict__ h2ib,
                                                 float* __restrict__ xcur) {
  __shared__ float red[512];
  int b = blockIdx.x, j = threadIdx.x;
  red[j] = hid[(size_t)b * H_DIM + j] * h2iw[j];
  __syncthreads();
  for (int s = 256; s > 0; s >>= 1) { if (j < s) red[j] += red[j + s]; __syncthreads(); }
  if (j == 0) xcur[b] = red[0] + h2ib[0];
}

__global__ void delta_kernel(const float* __restrict__ inputs, const float* __restrict__ est,
                             float* __restrict__ delta, _Float16* __restrict__ deltaH,
                             float* __restrict__ outIM) {
  int i = blockIdx.x * blockDim.x + threadIdx.x;
  if (i < B_SZ * T_LEN) {
    int b = i / T_LEN, t = i - b * T_LEN;
    float im = inputs[(size_t)b * 32 * T_LEN + t]; // channel 0
    float d  = im - est[i];
    delta[i] = d; deltaH[i] = (_Float16)d;
    outIM[i] = im;
  }
}

__global__ void final_kernel(const float* __restrict__ inputs, const float* __restrict__ g2,
                             const float* __restrict__ delta, float* __restrict__ out2) {
  int i = blockIdx.x * blockDim.x + threadIdx.x;
  if (i < B_SZ * T_LEN) {
    int b = i / T_LEN, t = i - b * T_LEN;
    float im = inputs[(size_t)b * 32 * T_LEN + t];
    float a  = 1.0f / (1.0f + __expf(-g2[i]));
    out2[i] = im - a * delta[i];
  }
}

// ---------------- host launcher ----------------
extern "C" void kernel_launch(void* const* d_in, const int* in_sizes, int n_in,
                              void* d_out, int out_size, void* d_ws, size_t ws_size,
                              hipStream_t stream) {
  (void)in_sizes; (void)n_in; (void)out_size; (void)ws_size;

  const float* inputs = (const float*)d_in[0];
  // tcn_params tree (None leaves dropped): layers 0-2 have 6 tensors, layer 3 has 4
  const float* c1w[4] = {(const float*)d_in[1],  (const float*)d_in[7],  (const float*)d_in[13], (const float*)d_in[19]};
  const float* c1b[4] = {(const float*)d_in[2],  (const float*)d_in[8],  (const float*)d_in[14], (const float*)d_in[20]};
  const float* c2w[4] = {(const float*)d_in[3],  (const float*)d_in[9],  (const float*)d_in[15], (const float*)d_in[21]};
  const float* c2b[4] = {(const float*)d_in[4],  (const float*)d_in[10], (const float*)d_in[16], (const float*)d_in[22]};
  const float* dwp[4] = {(const float*)d_in[5],  (const float*)d_in[11], (const float*)d_in[17], nullptr};
  const float* dbp[4] = {(const float*)d_in[6],  (const float*)d_in[12], (const float*)d_in[18], nullptr};
  const float* ar_w  = (const float*)d_in[23];
  const float* ar_b  = (const float*)d_in[24];
  const float* h2i_w = (const float*)d_in[25];
  const float* h2i_b = (const float*)d_in[26];
  const float* f_wih = (const float*)d_in[27];
  const float* f_whh = (const float*)d_in[28];
  const float* f_bih = (const float*)d_in[29];
  const float* f_bhh = (const float*)d_in[30];
  const float* f_fcw = (const float*)d_in[31];
  const float* f_fcb = (const float*)d_in[32];
  const float* e_wih = (const float*)d_in[33];
  const float* e_whh = (const float*)d_in[34];
  const float* e_bih = (const float*)d_in[35];
  const float* e_bhh = (const float*)d_in[36];
  const float* e_fcw = (const float*)d_in[37];
  const float* e_fcb = (const float*)d_in[38];
  const float* dag_w1 = (const float*)d_in[39];
  const float* dag_b1 = (const float*)d_in[40];
  const float* dag_w2 = (const float*)d_in[41];
  const float* dag_b2 = (const float*)d_in[42];

  // --- workspace carving ---
  size_t off = 0;
  auto carve = [&](size_t bytes) -> void* {
    off = (off + 255) & ~(size_t)255;
    void* p = (char*)d_ws + off;
    off += bytes;
    return p;
  };
  _Float16* inH    = (_Float16*)carve((size_t)B_SZ * 32 * T_LEN * 2);
  _Float16* buf0   = (_Float16*)carve((size_t)B_SZ * 512 * T_LEN * 2);
  _Float16* buf1   = (_Float16*)carve((size_t)B_SZ * 512 * T_LEN * 2);
  _Float16* buf2   = (_Float16*)carve((size_t)B_SZ * 512 * T_LEN * 2);
  float*    hidden = (float*)   carve((size_t)B_SZ * H_DIM * 4);
  _Float16* hidH   = (_Float16*)carve((size_t)B_SZ * H_DIM * 2);
  _Float16* whhF   = (_Float16*)carve((size_t)3 * H_DIM * H_DIM * 2);
  _Float16* whhE   = (_Float16*)carve((size_t)3 * H_DIM * H_DIM * 2);
  _Float16* arwH   = (_Float16*)carve((size_t)F_LEN * H_DIM * 2);
  _Float16* dw1H   = (_Float16*)carve((size_t)T_LEN * T_LEN * 2);
  _Float16* dw2H   = (_Float16*)carve((size_t)T_LEN * T_LEN * 2);
  float*    gh     = (float*)   carve((size_t)B_SZ * 3 * H_DIM * 4);
  float*    hbuf   = (float*)   carve((size_t)B_SZ * H_DIM * 4);
  _Float16* hH     = (_Float16*)carve((size_t)B_SZ * H_DIM * 2);
  float*    xcur   = (float*)   carve((size_t)B_SZ * 4);
  float*    delta  = (float*)   carve((size_t)B_SZ * T_LEN * 4);
  _Float16* deltaH = (_Float16*)carve((size_t)B_SZ * T_LEN * 2);
  float*    g1     = (float*)   carve((size_t)B_SZ * T_LEN * 4);
  _Float16* g1H    = (_Float16*)carve((size_t)B_SZ * T_LEN * 2);
  float*    g2     = (float*)   carve((size_t)B_SZ * T_LEN * 4);
  // repacked f16 conv weights [co][tap][ci]
  const int chans[4] = {128, 256, 512, 512};
  _Float16 *c1wH[4], *c2wH[4], *dwH[4];
  {
    int ci_ = 32;
    for (int L = 0; L < 4; ++L) {
      int co_ = chans[L];
      c1wH[L] = (_Float16*)carve((size_t)co_ * ci_ * 5 * 2);
      c2wH[L] = (_Float16*)carve((size_t)co_ * co_ * 5 * 2);
      dwH[L]  = dwp[L] ? (_Float16*)carve((size_t)co_ * ci_ * 2) : nullptr;
      ci_ = co_;
    }
  }

  // --- output regions (flat concat, f32) ---
  float* outForecast = (float*)d_out;                       // (B,1,96)
  float* out2        = (float*)d_out + 24576;               // (B,336,1)
  float* outAr       = (float*)d_out + 110592;              // (B,96,1)
  float* outEst      = (float*)d_out + 135168;              // (B,336,1)
  float* outIM       = (float*)d_out + 221184;              // (B,336,1)

  const int TB = 256;
  auto cvt = [&](const float* s, _Float16* d, int n) {
    cvt_f32_to_f16_kernel<<<(n + TB - 1) / TB, TB, 0, stream>>>(s, d, n);
  };
  cvt(inputs, inH, B_SZ * 32 * T_LEN);
  cvt(f_whh, whhF, 3 * H_DIM * H_DIM);
  cvt(e_whh, whhE, 3 * H_DIM * H_DIM);
  cvt(ar_w, arwH, F_LEN * H_DIM);
  cvt(dag_w1, dw1H, T_LEN * T_LEN);
  cvt(dag_w2, dw2H, T_LEN * T_LEN);

  auto repack = [&](const float* s, _Float16* d, int Cout_, int Cin_, int K_) {
    int n = Cout_ * Cin_ * K_;
    repack_w_kernel<<<(n + TB - 1) / TB, TB, 0, stream>>>(s, d, Cout_, Cin_, K_);
  };
  {
    int ci_ = 32;
    for (int L = 0; L < 4; ++L) {
      int co_ = chans[L];
      repack(c1w[L], c1wH[L], co_, ci_, 5);
      repack(c2w[L], c2wH[L], co_, co_, 5);
      if (dwp[L]) repack(dwp[L], dwH[L], co_, ci_, 1);
      ci_ = co_;
    }
  }

  // --- TCN encoder ---
  const _Float16* cur = inH;
  int ci = 32;
  dim3 cgrid(T_LEN / 16, B_SZ);
  auto conv = [&](const _Float16* xin, const _Float16* wp, const float* bp,
                  const _Float16* rp, _Float16* op, int Cin_, int Cout_,
                  int Ktaps_, int dil_, int mode_) {
    if (Ktaps_ == 1) {
      conv_wmma_kernel<1, 1><<<cgrid, 128, 0, stream>>>(xin, wp, bp, rp, op, Cin_, Cout_, mode_);
    } else {
      switch (dil_) {
        case 1: conv_wmma_kernel<5, 1><<<cgrid, 128, 0, stream>>>(xin, wp, bp, rp, op, Cin_, Cout_, mode_); break;
        case 2: conv_wmma_kernel<5, 2><<<cgrid, 128, 0, stream>>>(xin, wp, bp, rp, op, Cin_, Cout_, mode_); break;
        case 4: conv_wmma_kernel<5, 4><<<cgrid, 128, 0, stream>>>(xin, wp, bp, rp, op, Cin_, Cout_, mode_); break;
        default: conv_wmma_kernel<5, 8><<<cgrid, 128, 0, stream>>>(xin, wp, bp, rp, op, Cin_, Cout_, mode_); break;
      }
    }
  };
  for (int L = 0; L < 4; ++L) {
    const int co = chans[L];
    const int d  = 1 << L;
    const _Float16* resP;
    if (dwp[L]) {
      conv(cur, dwH[L], dbp[L], nullptr, buf1, ci, co, 1, 1, 0);
      resP = buf1;
    } else {
      resP = cur;
    }
    conv(cur, c1wH[L], c1b[L], nullptr, buf0, ci, co, 5, d, 1);
    _Float16* outP = (resP == cur) ? buf1 : buf2;
    conv(buf0, c2wH[L], c2b[L], resP, outP, co, co, 5, d, 2);
    cur = outP;
    ci = co;
  }

  extract_hidden_kernel<<<(B_SZ * H_DIM + TB - 1) / TB, TB, 0, stream>>>(cur, hidden, hidH);

  // --- AR head: (B,512) x (512,96) ---
  auto pair_grid = [&](int Ndim) { int pairs = (Ndim + 31) / 32; return dim3((pairs + 3) / 4, B_SZ / 16); };
  gemm_wmma_kernel<512><<<pair_grid(F_LEN), 128, 0, stream>>>(hidH, arwH, ar_b, outAr, nullptr,
                                                              B_SZ, F_LEN, 0);

  // --- forecast GRU (96 steps) ---
  f0_init_kernel<<<1, TB, 0, stream>>>(outAr, xcur);
  init_h_kernel<<<(B_SZ * H_DIM + TB - 1) / TB, TB, 0, stream>>>(hidden, hbuf, hH);
  dim3 ghgrid = pair_grid(3 * H_DIM);
  for (int s = 0; s < F_LEN; ++s) {
    gemm_wmma_kernel<512><<<ghgrid, 128, 0, stream>>>(hH, whhF, f_bhh, gh, nullptr,
                                                      B_SZ, 3 * H_DIM, 0);
    gru_gate_kernel<<<B_SZ, 512, 0, stream>>>(gh, f_wih, f_bih, f_fcw, f_fcb,
                                              hbuf, hH, xcur, outForecast, s, F_LEN);
  }

  // --- estimate GRU (336 steps) ---
  init_h_kernel<<<(B_SZ * H_DIM + TB - 1) / TB, TB, 0, stream>>>(hidden, hbuf, hH);
  e0_kernel<<<B_SZ, 512, 0, stream>>>(hidden, h2i_w, h2i_b, xcur);
  for (int s = 0; s < T_LEN; ++s) {
    gemm_wmma_kernel<512><<<ghgrid, 128, 0, stream>>>(hH, whhE, e_bhh, gh, nullptr,
                                                      B_SZ, 3 * H_DIM, 0);
    gru_gate_kernel<<<B_SZ, 512, 0, stream>>>(gh, e_wih, e_bih, e_fcw, e_fcb,
                                              hbuf, hH, xcur, outEst, s, T_LEN);
  }

  // --- DAG head ---
  delta_kernel<<<(B_SZ * T_LEN + TB - 1) / TB, TB, 0, stream>>>(inputs, outEst,
                                                                delta, deltaH, outIM);
  gemm_wmma_kernel<336><<<pair_grid(T_LEN), 128, 0, stream>>>(deltaH, dw1H, dag_b1, g1, g1H,
                                                              B_SZ, T_LEN, 1);
  gemm_wmma_kernel<336><<<pair_grid(T_LEN), 128, 0, stream>>>(g1H, dw2H, dag_b2, g2, nullptr,
                                                              B_SZ, T_LEN, 0);
  final_kernel<<<(B_SZ * T_LEN + TB - 1) / TB, TB, 0, stream>>>(inputs, g2, delta, out2);
}
